// CausalSelfAttention_58196806861045
// MI455X (gfx1250) — matrix operand order repaired and checked by reading, not compiled, not run on hardware
//
#include <hip/hip_runtime.h>
#include <hip/hip_bf16.h>

// ---------------------------------------------------------------------------
// Causal self-attention forward for MI455X (gfx1250, wave32, WMMA).
// fp32 inputs -> bf16, all matmuls via v_wmma_f32_16x16x32_bf16 (f32 accum),
// softmax in f32. GEMMs use double-buffered LDS software pipeline; attention
// pipelines V (register prefetch) and K (parity double-buffer) loads.
// ---------------------------------------------------------------------------

#define BATCH 4
#define SEQ   2048
#define DMODEL 1024
#define NHEAD 16
#define HDIM  64

typedef __attribute__((ext_vector_type(16))) __bf16 v16bf;
typedef __attribute__((ext_vector_type(8)))  float  v8f;

union Frag {                 // one WMMA 16-bit A/B operand (8 VGPRs)
  v16bf v;
  unsigned short h[16];
  uint4 q[2];
};
union F8 {                   // one WMMA f32 C/D operand (8 VGPRs)
  v8f v;
  float f[8];
};

__device__ __forceinline__ unsigned short f2bf(float f) {
  union { float f; unsigned int u; } cv; cv.f = f;
  unsigned int u = cv.u;
  u += 0x7fffu + ((u >> 16) & 1u);          // round-to-nearest-even
  return (unsigned short)(u >> 16);
}

__device__ __forceinline__ v8f wmma_bf16(v16bf a, v16bf b, v8f c) {
  return __builtin_amdgcn_wmma_f32_16x16x32_bf16(
      /*neg_a=*/false, a, /*neg_b=*/false, b,
      /*c_mod=*/(short)0, c, /*reuse_a=*/false, /*reuse_b=*/false);
}

// ---------------------------------------------------------------------------
// Kernel 0: fp32 -> bf16 (vectorized x4, grid-stride)
// ---------------------------------------------------------------------------
__global__ void cvt_f32_bf16(const float* __restrict__ src,
                             unsigned short* __restrict__ dst, int n4) {
  int i = blockIdx.x * blockDim.x + threadIdx.x;
  int stride = gridDim.x * blockDim.x;
  for (; i < n4; i += stride) {
    float4 f = ((const float4*)src)[i];
    ushort4 o;
    o.x = f2bf(f.x); o.y = f2bf(f.y); o.z = f2bf(f.z); o.w = f2bf(f.w);
    ((ushort4*)dst)[i] = o;
  }
}

// ---------------------------------------------------------------------------
// Kernel 1/3: bf16 GEMM  C(MxN) = A(MxK) * B(KxN) + bias
//   block tile 128x128x32, 256 threads = 8 waves, wave tile 32x64 (2x4 WMMAs)
//   double-buffered LDS, one barrier per K-step, global loads overlapped
//   with WMMA of the current tile.
//   mode 0: fp32 row-major output (projection)
//   mode 1: bf16 scatter into QKV [3][B][H][T][HD] layout (QKV projection)
// ---------------------------------------------------------------------------
__global__ __launch_bounds__(256) void gemm_bf16(
    const unsigned short* __restrict__ A,   // M x K, bf16
    const unsigned short* __restrict__ Bw,  // K x N, bf16
    const float* __restrict__ bias,         // N
    float* __restrict__ outF,               // mode 0
    unsigned short* __restrict__ outQKV,    // mode 1
    int M, int N, int K, int mode) {
  __shared__ unsigned short As[2][128][40]; // row-major A tile (pad -> 16B align)
  __shared__ unsigned short Bt[2][128][40]; // TRANSPOSED B tile: Bt[n][k]

  const int tid    = threadIdx.x;
  const int wave   = tid >> 5;
  const int lane   = tid & 31;
  const int lane16 = lane & 15;
  const int half   = lane >> 4;
  const int wm     = wave >> 1;             // 0..3
  const int wn     = wave & 1;              // 0..1
  const int m0     = blockIdx.y * 128;
  const int n0     = blockIdx.x * 128;

  F8 acc[2][4];
#pragma unroll
  for (int mi = 0; mi < 2; mi++)
#pragma unroll
    for (int ni = 0; ni < 4; ni++)
#pragma unroll
      for (int r = 0; r < 8; r++) acc[mi][ni].f[r] = 0.f;

  const int arow = tid >> 1, ahalf = tid & 1;      // A loader: 2 thr/row
  const int brow = tid >> 3, bseg  = tid & 7;      // B loader: 8 thr/row
  const unsigned short* aptr = A  + (size_t)(m0 + arow) * K + 16 * ahalf;
  const unsigned short* bptr = Bw + (size_t)brow * N + n0 + 16 * bseg;
  const int KT = K >> 5;                    // K-steps of 32

  // ---- prologue: stage tile 0 into LDS buffer 0 ----
  {
    const uint4* ap = (const uint4*)aptr;
    uint4 ra0 = ap[0], ra1 = ap[1];
    Frag rb;
    const uint4* bp = (const uint4*)bptr;
    rb.q[0] = bp[0]; rb.q[1] = bp[1];
    *(uint4*)&As[0][arow][16 * ahalf]     = ra0;
    *(uint4*)&As[0][arow][16 * ahalf + 8] = ra1;
#pragma unroll
    for (int j = 0; j < 16; j++) Bt[0][16 * bseg + j][brow] = rb.h[j];
  }
  __syncthreads();

  for (int kt = 0; kt < KT; kt++) {
    const int cur = kt & 1, nxt = cur ^ 1;
    const bool more = (kt + 1 < KT);

    // ---- issue global loads for tile kt+1 (latency hidden by WMMAs) ----
    uint4 ra0, ra1;
    Frag rb;
    if (more) {
      const uint4* ap = (const uint4*)(aptr + (size_t)(kt + 1) * 32);
      ra0 = ap[0]; ra1 = ap[1];
      const uint4* bp = (const uint4*)(bptr + (size_t)(kt + 1) * 32 * N);
      rb.q[0] = bp[0]; rb.q[1] = bp[1];
    }
    if (kt + 2 < KT) {                      // global_prefetch_b8 two tiles ahead
      __builtin_prefetch(aptr + (size_t)(kt + 2) * 32, 0, 1);
      __builtin_prefetch(bptr + (size_t)(kt + 2) * 32 * N, 0, 1);
    }

    // ---- gather all fragments from LDS[cur], then 8 back-to-back WMMAs ----
    Frag af[2];
#pragma unroll
    for (int mi = 0; mi < 2; mi++) {
      int row = 32 * wm + 16 * mi + lane16;
      af[mi].q[0] = *(const uint4*)&As[cur][row][16 * half];      // K = 16h+0..7
      af[mi].q[1] = *(const uint4*)&As[cur][row][16 * half + 8];  // K = 16h+8..15
    }
    Frag bfr[4];
#pragma unroll
    for (int ni = 0; ni < 4; ni++) {
      int col = 64 * wn + 16 * ni + lane16;
      bfr[ni].q[0] = *(const uint4*)&Bt[cur][col][16 * half];     // K = 16h+0..7
      bfr[ni].q[1] = *(const uint4*)&Bt[cur][col][16 * half + 8]; // K = 16h+8..15
    }
#pragma unroll
    for (int ni = 0; ni < 4; ni++)
#pragma unroll
      for (int mi = 0; mi < 2; mi++)
        acc[mi][ni].v = wmma_bf16(af[mi].v, bfr[ni].v, acc[mi][ni].v);

    // ---- store tile kt+1 into LDS[nxt] ----
    if (more) {
      *(uint4*)&As[nxt][arow][16 * ahalf]     = ra0;
      *(uint4*)&As[nxt][arow][16 * ahalf + 8] = ra1;
#pragma unroll
      for (int j = 0; j < 16; j++) Bt[nxt][16 * bseg + j][brow] = rb.h[j];
    }
    __syncthreads();
  }

  // ---- epilogue ----
#pragma unroll
  for (int mi = 0; mi < 2; mi++) {
#pragma unroll
    for (int ni = 0; ni < 4; ni++) {
      int col = n0 + 64 * wn + 16 * ni + lane16;
      float bv = bias[col];
#pragma unroll
      for (int r = 0; r < 8; r++) {
        int row = m0 + 32 * wm + 16 * mi + r + 8 * half;
        float val = acc[mi][ni].f[r] + bv;
        if (mode == 0) {
          outF[(size_t)row * N + col] = val;
        } else {
          int sec = col >> 10, cm = col & 1023;     // 0=Q 1=K 2=V
          int h = cm >> 6, hd = cm & 63;
          int b = row >> 11, t = row & 2047;
          size_t dst = ((((size_t)sec * BATCH + b) * NHEAD + h) * SEQ + t) * HDIM + hd;
          outQKV[dst] = f2bf(val);
        }
      }
    }
  }
}

// ---------------------------------------------------------------------------
// Kernel 2: flash attention forward. 1 WG = 1 (b,h) head x 128 query rows.
//   8 waves, each owns 16 query rows. V prefetched into registers one block
//   ahead; K fragments parity-pipelined across the 8 S-tiles.
// ---------------------------------------------------------------------------
__global__ __launch_bounds__(256) void attn_fwd(
    const unsigned short* __restrict__ Qb,  // [B][H][T][HD] bf16
    const unsigned short* __restrict__ Kb,
    const unsigned short* __restrict__ Vb,
    unsigned short* __restrict__ Yb) {      // [B][T][D] bf16
  __shared__ unsigned short Vt[64][136];    // V block transposed: Vt[hd][key]
  __shared__ unsigned short Ps[128][136];   // P bf16 (row band per wave)

  const int tid    = threadIdx.x;
  const int wave   = tid >> 5;
  const int lane   = tid & 31;
  const int lane16 = lane & 15;
  const int half   = lane >> 4;
  const int qb     = blockIdx.x;            // 0..15 query block
  const int bh     = blockIdx.y;            // 0..63 (b*H + h)
  const int b      = bh >> 4, h = bh & 15;
  const int q0     = qb * 128;
  const size_t headBase = (size_t)bh * SEQ * HDIM;

  // Q fragments for this wave's 16 rows (A-frag layout, straight from global)
  Frag qf[2];
  {
    const unsigned short* qp = Qb + headBase + (size_t)(q0 + 16 * wave + lane16) * HDIM;
#pragma unroll
    for (int ks = 0; ks < 2; ks++) {
      qf[ks].q[0] = *(const uint4*)(qp + 32 * ks + 16 * half);
      qf[ks].q[1] = *(const uint4*)(qp + 32 * ks + 16 * half + 8);
    }
  }

  F8 o[4];
  float mrow[8], lrow[8];
#pragma unroll
  for (int ht = 0; ht < 4; ht++)
#pragma unroll
    for (int r = 0; r < 8; r++) o[ht].f[r] = 0.f;
#pragma unroll
  for (int r = 0; r < 8; r++) { mrow[r] = -1e30f; lrow[r] = 0.f; }

  // ---- V register prefetch: block 0 ----
  const int vkey = tid >> 1, vseg = tid & 1;
  const unsigned short* vbase = Vb + headBase + (size_t)vkey * HDIM + 32 * vseg;
  Frag vr0, vr1;
  {
    const uint4* vp = (const uint4*)vbase;
    vr0.q[0] = vp[0]; vr0.q[1] = vp[1];
    vr1.q[0] = vp[2]; vr1.q[1] = vp[3];
  }

  for (int kb = 0; kb <= qb; kb++) {        // causal: skip blocks right of diag
    const int k0 = kb * 128;
    const bool diag = (kb == qb);
    __syncthreads();                        // prior PV reads of Vt complete
    // ---- scatter prefetched V block into LDS (transposed) ----
#pragma unroll
    for (int j = 0; j < 16; j++) Vt[32 * vseg + j][vkey]      = vr0.h[j];
#pragma unroll
    for (int j = 0; j < 16; j++) Vt[32 * vseg + 16 + j][vkey] = vr1.h[j];
    // ---- issue loads for next V block (hidden behind this block's math) ----
    if (kb < qb) {
      const uint4* vp = (const uint4*)(vbase + (size_t)(k0 + 128) * HDIM);
      vr0.q[0] = vp[0]; vr0.q[1] = vp[1];
      vr1.q[0] = vp[2]; vr1.q[1] = vp[3];
    }
    __syncthreads();

    // ---- S = Q K^T (per wave: 16 x 128), K frags pipelined by parity ----
    F8 s[8];
    Frag kf[2][2];
    const unsigned short* kbase = Kb + headBase + (size_t)(k0 + lane16) * HDIM;
    {
      const unsigned short* kp = kbase;     // nt = 0
      kf[0][0].q[0] = *(const uint4*)(kp + 16 * half);
      kf[0][0].q[1] = *(const uint4*)(kp + 16 * half + 8);
      kf[0][1].q[0] = *(const uint4*)(kp + 32 + 16 * half);
      kf[0][1].q[1] = *(const uint4*)(kp + 32 + 16 * half + 8);
    }
#pragma unroll
    for (int nt = 0; nt < 8; nt++) {
      const int p = nt & 1;
      if (nt < 7) {                         // prefetch K frags for nt+1
        const unsigned short* kp = kbase + (size_t)(nt + 1) * 16 * HDIM;
        kf[p ^ 1][0].q[0] = *(const uint4*)(kp + 16 * half);
        kf[p ^ 1][0].q[1] = *(const uint4*)(kp + 16 * half + 8);
        kf[p ^ 1][1].q[0] = *(const uint4*)(kp + 32 + 16 * half);
        kf[p ^ 1][1].q[1] = *(const uint4*)(kp + 32 + 16 * half + 8);
      }
      F8 sv;
#pragma unroll
      for (int r = 0; r < 8; r++) sv.f[r] = 0.f;
      sv.v = wmma_bf16(qf[0].v, kf[p][0].v, sv.v);
      sv.v = wmma_bf16(qf[1].v, kf[p][1].v, sv.v);
      s[nt] = sv;
    }
    // ---- scale + causal mask ----
#pragma unroll
    for (int nt = 0; nt < 8; nt++)
#pragma unroll
      for (int r = 0; r < 8; r++) {
        float v = s[nt].f[r] * 0.125f;      // 1/sqrt(64)
        if (diag) {
          int qr = 16 * wave + r + 8 * half;
          int kc = 16 * nt + lane16;
          if (kc > qr) v = -1e30f;
        }
        s[nt].f[r] = v;
      }
    // ---- online softmax (row lives in one 16-lane half) ----
#pragma unroll
    for (int r = 0; r < 8; r++) {
      float rm = -1e30f;
#pragma unroll
      for (int nt = 0; nt < 8; nt++) rm = fmaxf(rm, s[nt].f[r]);
#pragma unroll
      for (int d = 1; d < 16; d <<= 1) rm = fmaxf(rm, __shfl_xor(rm, d, 32));
      float mnew  = fmaxf(mrow[r], rm);
      float alpha = __expf(mrow[r] - mnew);
      mrow[r] = mnew;
      float rs = 0.f;
#pragma unroll
      for (int nt = 0; nt < 8; nt++) {
        float p = __expf(s[nt].f[r] - mnew);
        s[nt].f[r] = p;
        rs += p;
      }
#pragma unroll
      for (int d = 1; d < 16; d <<= 1) rs += __shfl_xor(rs, d, 32);
      lrow[r] = lrow[r] * alpha + rs;
#pragma unroll
      for (int ht = 0; ht < 4; ht++) o[ht].f[r] *= alpha;
    }
    // ---- P -> LDS bf16 (wave-private row band, no WG barrier needed) ----
#pragma unroll
    for (int nt = 0; nt < 8; nt++)
#pragma unroll
      for (int r = 0; r < 8; r++)
        Ps[16 * wave + r + 8 * half][16 * nt + lane16] = f2bf(s[nt].f[r]);

    // ---- O += P @ V : per K-step, 4 batched ds frag loads + 4 WMMAs ----
    Frag pf[4];
#pragma unroll
    for (int ks = 0; ks < 4; ks++) {
      int row = 16 * wave + lane16;
      pf[ks].q[0] = *(const uint4*)&Ps[row][32 * ks + 16 * half];
      pf[ks].q[1] = *(const uint4*)&Ps[row][32 * ks + 16 * half + 8];
    }
#pragma unroll
    for (int ks = 0; ks < 4; ks++) {
      Frag vfr[4];
#pragma unroll
      for (int ht = 0; ht < 4; ht++) {
        int hd = 16 * ht + lane16;
        vfr[ht].q[0] = *(const uint4*)&Vt[hd][32 * ks + 16 * half];
        vfr[ht].q[1] = *(const uint4*)&Vt[hd][32 * ks + 16 * half + 8];
      }
#pragma unroll
      for (int ht = 0; ht < 4; ht++)
        o[ht].v = wmma_bf16(pf[ks].v, vfr[ht].v, o[ht].v);
    }
  }

  // ---- finalize: O /= l, write Y as [B][T][D] bf16 ----
#pragma unroll
  for (int ht = 0; ht < 4; ht++)
#pragma unroll
    for (int r = 0; r < 8; r++) {
      int trow = q0 + 16 * wave + r + 8 * half;
      float val = o[ht].f[r] / lrow[r];
      size_t dst = ((size_t)b * SEQ + trow) * DMODEL + h * HDIM + 16 * ht + lane16;
      Yb[dst] = f2bf(val);
    }
}

// ---------------------------------------------------------------------------
// Host launch
// ---------------------------------------------------------------------------
extern "C" void kernel_launch(void* const* d_in, const int* in_sizes, int n_in,
                              void* d_out, int out_size, void* d_ws, size_t ws_size,
                              hipStream_t stream) {
  (void)in_sizes; (void)n_in; (void)out_size; (void)ws_size;
  const float* x      = (const float*)d_in[0];
  const float* w_attn = (const float*)d_in[1];
  const float* b_attn = (const float*)d_in[2];
  const float* w_proj = (const float*)d_in[3];
  const float* b_proj = (const float*)d_in[4];
  float* out = (float*)d_out;

  const size_t MROWS = (size_t)BATCH * SEQ;      // 8192
  unsigned short* Xb   = (unsigned short*)d_ws;              // 8192x1024
  unsigned short* Wb   = Xb  + MROWS * DMODEL;               // 1024x3072
  unsigned short* Wpb  = Wb  + (size_t)DMODEL * 3 * DMODEL;  // 1024x1024
  unsigned short* QKVb = Wpb + (size_t)DMODEL * DMODEL;      // 3 x [B][H][T][HD]
  unsigned short* Yb   = QKVb + 3 * MROWS * DMODEL;          // 8192x1024
  unsigned short* Qp = QKVb;
  unsigned short* Kp = QKVb + MROWS * DMODEL;
  unsigned short* Vp = QKVb + 2 * MROWS * DMODEL;

  cvt_f32_bf16<<<1024, 256, 0, stream>>>(x,      Xb,  (int)(MROWS * DMODEL / 4));
  cvt_f32_bf16<<<1024, 256, 0, stream>>>(w_attn, Wb,  (int)(DMODEL * 3 * DMODEL / 4));
  cvt_f32_bf16<<<1024, 256, 0, stream>>>(w_proj, Wpb, (int)(DMODEL * DMODEL / 4));

  // QKV projection: 8192x3072 = x @ w_attn + b_attn, scattered to head layout
  gemm_bf16<<<dim3(3 * DMODEL / 128, MROWS / 128), 256, 0, stream>>>(
      Xb, Wb, b_attn, nullptr, QKVb, (int)MROWS, 3 * DMODEL, DMODEL, 1);

  // Flash attention: grid = (T/128 q-blocks, B*H heads)
  attn_fwd<<<dim3(SEQ / 128, BATCH * NHEAD), 256, 0, stream>>>(Qp, Kp, Vp, Yb);

  // Output projection: out = Y @ w_proj + b_proj (fp32)
  gemm_bf16<<<dim3(DMODEL / 128, MROWS / 128), 256, 0, stream>>>(
      Yb, Wpb, b_proj, out, nullptr, (int)MROWS, DMODEL, DMODEL, 0);
}
